// GDAModule_cls_32856499814842
// MI455X (gfx1250) — compile-verified, matrix operand order checked by reading, not run on hardware
//
#include <hip/hip_runtime.h>
#include <hip/hip_bf16.h>

// ---------------------------------------------------------------------------
// GDANet forward for MI455X (gfx1250, wave32, WMMA bf16 16x16x32).
// LDS tiles are row-major-in-K with 16B-aligned rows: WMMA fragments are two
// ds_load_b128 per lane; staging uses straight-line 16-element runs per thread
// (batched global loads, one wait, two ds_store_b128).
// ---------------------------------------------------------------------------

typedef __attribute__((ext_vector_type(16))) __bf16 v16bf;
typedef __attribute__((ext_vector_type(8)))  __bf16 v8bf;
typedef __attribute__((ext_vector_type(8)))  float  v8f;

#define BATCH 8
#define NPTS  2048
#define KNN_K 20
#define KG    64
#define MPTS  512
#define BN_EPS 1e-5f
#define CMAX 132

// LDS pitches (bf16 elements); pitch*2 bytes must be a multiple of 16.
#define LDA_G  40    // gemm tiles: 40*2 = 80 B pitch
#define LDA_E  136   // edge tiles: 136*2 = 272 B pitch

static __device__ inline v8f zero_v8f() {
  v8f z;
#pragma unroll
  for (int i = 0; i < 8; ++i) z[i] = 0.f;
  return z;
}

// Pack 16 floats -> two v8bf and store contiguously (both stores 16B aligned).
static __device__ inline void store_run16(__bf16* d, const float* va) {
  v8bf lo, hi;
#pragma unroll
  for (int e = 0; e < 8; ++e) { lo[e] = (__bf16)va[e]; hi[e] = (__bf16)va[e + 8]; }
  *(v8bf*)(d)     = lo;
  *(v8bf*)(d + 8) = hi;
}

// Fragment from LDS tile laid out [row][K] with pitch ld (row = M for A, N for
// transposed-B). Per ISA 7.12.2: lanes 0-15 -> K={0..7,16..23},
// lanes 16-31 -> K={8..15,24..31}. Both runs are 16B-aligned contiguous.
static __device__ inline v16bf load_frag_rowk(const __bf16* s, int ld, int row0,
                                              int k0, int lane) {
  const __bf16* p = s + (row0 + (lane & 15)) * ld + k0 + ((lane >> 4) << 3);
  v8bf lo = *(const v8bf*)(p);
  v8bf hi = *(const v8bf*)(p + 16);
  return __builtin_shufflevector(lo, hi, 0, 1, 2, 3, 4, 5, 6, 7,
                                 8, 9, 10, 11, 12, 13, 14, 15);
}

static __device__ inline v8f wmma_bf16(v16bf a, v16bf b, v8f c) {
  return __builtin_amdgcn_wmma_f32_16x16x32_bf16(false, a, false, b, (short)0, c,
                                                 false, false);
}

// ---------------------------------------------------------------------------
// Generic strided GEMM with WMMA:  C[b] = epi( alpha * A[b](MxK) * B[b](KxN) )
// epi: v = alpha*acc; if(scale) v*=scale[m]; if(shift) v+=shift[m];
//      if(relu) v=max(v,0); if(resid) v+=resid[b,m,n];  store to C.
// 64x64 tile per 128-thread workgroup (4 waves, 32x32 quadrant each).
// ---------------------------------------------------------------------------
__global__ __launch_bounds__(128) void gemm_wmma(
    const float* __restrict__ A, long long ars, long long acs, long long abs_,
    const float* __restrict__ Bm, long long brs, long long bcs, long long bbs,
    float* __restrict__ C, long long crs, long long cbs,
    const float* __restrict__ resid, long long rrs, long long rbs,
    const float* __restrict__ scale, const float* __restrict__ shift,
    int M, int N, int Kd, float alpha, int relu)
{
  __shared__ __align__(16) __bf16 sA [64 * LDA_G];  // [m][k]
  __shared__ __align__(16) __bf16 sBt[64 * LDA_G];  // [n][k] (B transposed)
  int b  = blockIdx.z;
  int n0 = blockIdx.x * 64;
  int m0 = blockIdx.y * 64;
  int tid = threadIdx.x, lane = tid & 31, wv = tid >> 5;
  int mq = (wv >> 1) * 32;
  int nq = (wv & 1) * 32;
  const float* Ab = A + (long long)b * abs_;
  const float* Bb = Bm + (long long)b * bbs;
  v8f acc00 = zero_v8f(), acc01 = zero_v8f(), acc10 = zero_v8f(), acc11 = zero_v8f();
  for (int k0 = 0; k0 < Kd; k0 += 32) {
    if (k0 + 32 <= Kd) {
      // ---- fast staging: one 16-element run per thread, no per-element predication
      {
        int r  = tid >> 1;
        int kh = (tid & 1) << 4;
        float va[16];
        if (m0 + r < M) {
          const float* pa = Ab + (long long)(m0 + r) * ars + (long long)(k0 + kh) * acs;
          if (acs == 1 && (ars & 3) == 0) {
            const float4* p4 = (const float4*)pa;
#pragma unroll
            for (int q = 0; q < 4; ++q) {
              float4 t4 = p4[q];
              va[4*q+0] = t4.x; va[4*q+1] = t4.y; va[4*q+2] = t4.z; va[4*q+3] = t4.w;
            }
          } else {
#pragma unroll
            for (int e = 0; e < 16; ++e) va[e] = pa[(long long)e * acs];
          }
        } else {
#pragma unroll
          for (int e = 0; e < 16; ++e) va[e] = 0.f;
        }
        store_run16(&sA[r * LDA_G + kh], va);
      }
      {
        int c  = tid & 63;
        int kh = (tid >> 6) << 4;
        float vb[16];
        if (n0 + c < N) {
          const float* pb = Bb + (long long)(k0 + kh) * brs + (long long)(n0 + c) * bcs;
          if (brs == 1 && (bcs & 3) == 0) {
            const float4* p4 = (const float4*)pb;
#pragma unroll
            for (int q = 0; q < 4; ++q) {
              float4 t4 = p4[q];
              vb[4*q+0] = t4.x; vb[4*q+1] = t4.y; vb[4*q+2] = t4.z; vb[4*q+3] = t4.w;
            }
          } else {
#pragma unroll
            for (int e = 0; e < 16; ++e) vb[e] = pb[(long long)e * brs];
          }
        } else {
#pragma unroll
          for (int e = 0; e < 16; ++e) vb[e] = 0.f;
        }
        store_run16(&sBt[c * LDA_G + kh], vb);
      }
    } else {
      // ---- K-tail: guarded scalar staging
      for (int i = tid; i < 64 * 32; i += 128) {
        int r = i >> 5, c = i & 31;
        int gm = m0 + r, gk = k0 + c;
        float v = (gm < M && gk < Kd) ? Ab[(long long)gm * ars + (long long)gk * acs] : 0.f;
        sA[r * LDA_G + c] = (__bf16)v;
      }
      for (int i = tid; i < 32 * 64; i += 128) {
        int r = i >> 6, c = i & 63;          // r = k, c = n
        int gk = k0 + r, gn = n0 + c;
        float v = (gk < Kd && gn < N) ? Bb[(long long)gk * brs + (long long)gn * bcs] : 0.f;
        sBt[c * LDA_G + r] = (__bf16)v;
      }
    }
    __syncthreads();
    v16bf a0 = load_frag_rowk(sA,  LDA_G, mq,      0, lane);
    v16bf a1 = load_frag_rowk(sA,  LDA_G, mq + 16, 0, lane);
    v16bf b0 = load_frag_rowk(sBt, LDA_G, nq,      0, lane);
    v16bf b1 = load_frag_rowk(sBt, LDA_G, nq + 16, 0, lane);
    acc00 = wmma_bf16(a0, b0, acc00);
    acc01 = wmma_bf16(a0, b1, acc01);
    acc10 = wmma_bf16(a1, b0, acc10);
    acc11 = wmma_bf16(a1, b1, acc11);
    __syncthreads();
  }
  float* Cb = C + (long long)b * cbs;
  const float* Rb = resid ? resid + (long long)b * rbs : (const float*)0;
  int lcol = lane & 15, lrow = (lane >> 4) * 8;
#pragma unroll
  for (int ti = 0; ti < 2; ++ti) {
#pragma unroll
    for (int tj = 0; tj < 2; ++tj) {
      v8f acc = (ti == 0) ? (tj == 0 ? acc00 : acc01) : (tj == 0 ? acc10 : acc11);
      int gn = n0 + nq + tj * 16 + lcol;
      if (gn >= N) continue;
#pragma unroll
      for (int r = 0; r < 8; ++r) {
        int gm = m0 + mq + ti * 16 + lrow + r;
        if (gm >= M) continue;
        float v = acc[r] * alpha;
        if (scale) v = scale[gm] * v;
        if (shift) v += shift[gm];
        if (relu)  v = fmaxf(v, 0.f);
        if (Rb)    v += Rb[(long long)gm * rrs + gn];
        Cb[(long long)gm * crs + gn] = v;
      }
    }
  }
}

// ---------------------------------------------------------------------------
// Fused edge-conv layer2: columns are (point, neighbor) pairs.
// h1[c,col] = relu(s1[c]*(P[c,nbr]+Q[c,n]) + t1[c]) is produced in LDS (bf16,
// stored [col][c] so fragments are contiguous-in-K), then
// out[b,o,n] = atomic-max over k of relu(s2[o]*(W2*h1) + t2[o]).
// K1 and O2 are multiples of 16 (64 or 128).
// ---------------------------------------------------------------------------
__global__ __launch_bounds__(128) void edge2_wmma(
    const float* __restrict__ P, const float* __restrict__ Q,
    const int* __restrict__ idx,
    const float* __restrict__ s1, const float* __restrict__ t1,
    const float* __restrict__ W2,
    const float* __restrict__ s2, const float* __restrict__ t2,
    float* __restrict__ outp, int N_, int K1, int O2, int knb)
{
  __shared__ __align__(16) __bf16 sW [128 * LDA_E];  // [o][c]
  __shared__ __align__(16) __bf16 sHt[64 * LDA_E];   // [col][c]
  __shared__ int sN[64];
  __shared__ int sNb[64];
  int b = blockIdx.y;
  long long col0 = (long long)blockIdx.x * 64;
  long long totalCols = (long long)N_ * knb;
  int tid = threadIdx.x, lane = tid & 31, wv = tid >> 5;
  // stage W2 as 16-element vector runs (K1 multiple of 16, W2 16B aligned)
  for (int i0 = tid * 16; i0 < O2 * K1; i0 += 128 * 16) {
    int o = i0 / K1, c = i0 - o * K1;
    float va[16];
    const float4* p4 = (const float4*)(W2 + i0);
#pragma unroll
    for (int q = 0; q < 4; ++q) {
      float4 t4 = p4[q];
      va[4*q+0] = t4.x; va[4*q+1] = t4.y; va[4*q+2] = t4.z; va[4*q+3] = t4.w;
    }
    store_run16(&sW[o * LDA_E + c], va);
  }
  if (tid < 64) {
    long long cg = col0 + tid;
    if (cg < totalCols) {
      int n = (int)(cg / knb);
      sN[tid]  = n;
      sNb[tid] = idx[((long long)b * N_ + n) * knb + (int)(cg - (long long)n * knb)];
    } else { sN[tid] = -1; sNb[tid] = 0; }
  }
  __syncthreads();
  const float* Pb = P + (long long)b * K1 * N_;
  const float* Qb = Q + (long long)b * K1 * N_;
  // h1 producer: one (column, 16-channel run) task per thread iteration.
  int tasks = 64 * (K1 >> 4);
  for (int t = tid; t < tasks; t += 128) {
    int cl  = t & 63;
    int ch0 = (t >> 6) << 4;
    int n = sN[cl];
    float va[16];
    if (n >= 0) {
      int nb = sNb[cl];
#pragma unroll
      for (int e = 0; e < 16; ++e) {
        float z = Pb[(long long)(ch0 + e) * N_ + nb] + Qb[(long long)(ch0 + e) * N_ + n];
        va[e] = fmaxf(s1[ch0 + e] * z + t1[ch0 + e], 0.f);
      }
    } else {
#pragma unroll
      for (int e = 0; e < 16; ++e) va[e] = 0.f;
    }
    store_run16(&sHt[cl * LDA_E + ch0], va);
  }
  __syncthreads();
  int colTile = wv * 16;
  int lcol = lane & 15, lrow = (lane >> 4) * 8;
  int cl = colTile + lcol;
  int n  = sN[cl];
  float* Ob = outp + (long long)b * O2 * N_;
  for (int rt = 0; rt < O2; rt += 16) {
    v8f acc = zero_v8f();
    for (int k0 = 0; k0 < K1; k0 += 32) {
      v16bf a  = load_frag_rowk(sW,  LDA_E, rt,      k0, lane);
      v16bf bb = load_frag_rowk(sHt, LDA_E, colTile, k0, lane);
      acc = wmma_bf16(a, bb, acc);
    }
    if (n >= 0) {
#pragma unroll
      for (int r = 0; r < 8; ++r) {
        int o = rt + lrow + r;
        float v = fmaxf(s2[o] * acc[r] + t2[o], 0.f);
        atomicMax((int*)&Ob[(long long)o * N_ + n], __float_as_int(v)); // v >= 0
      }
    }
  }
}

// ---------------------------------------------------------------------------
// kNN top-k (includes self). One thread per query point; candidate tiles in
// LDS; sorted insertion (stable: ties keep smallest index first).
// ---------------------------------------------------------------------------
__global__ __launch_bounds__(256) void knn_kernel(
    const float* __restrict__ X, int C, int N_, int k, int* __restrict__ idx,
    long long xbs)
{
  __shared__ float sP[32][CMAX];
  int b = blockIdx.y;
  int n = blockIdx.x * 256 + threadIdx.x;
  const float* Xb = X + (long long)b * xbs;
  float q[CMAX];
  float bv[KG]; int bi[KG];
  if (n < N_)
    for (int c = 0; c < C; ++c) q[c] = Xb[(long long)c * N_ + n];
  for (int j = 0; j < KG; ++j) { bv[j] = -3.4e38f; bi[j] = 0; }
  for (int t0 = 0; t0 < N_; t0 += 32) {
    __syncthreads();
    for (int i = threadIdx.x; i < 32 * C; i += 256) {
      int cand = i / C, ch = i - cand * C;
      sP[cand][ch] = Xb[(long long)ch * N_ + (t0 + cand)];
    }
    __syncthreads();
    if (n < N_) {
      for (int cand = 0; cand < 32; ++cand) {
        float d2 = 0.f;
        for (int c = 0; c < C; ++c) { float d = q[c] - sP[cand][c]; d2 += d * d; }
        float v = -d2;
        if (v > bv[k - 1]) {
          int pos = k - 1;
          while (pos > 0 && bv[pos - 1] < v) {
            bv[pos] = bv[pos - 1]; bi[pos] = bi[pos - 1]; --pos;
          }
          bv[pos] = v; bi[pos] = t0 + cand;
        }
      }
    }
  }
  if (n < N_)
    for (int j = 0; j < k; ++j) idx[((long long)b * N_ + n) * k + j] = bi[j];
}

// GDM high-pass score: Gaussian affinity over 64-NN, thresholded low-pass.
__global__ __launch_bounds__(256) void gdm_score(
    const float* __restrict__ X, const int* __restrict__ knn64,
    float* __restrict__ score, int N_)
{
  int b = blockIdx.y;
  int n = blockIdx.x * 256 + threadIdx.x;
  if (n >= N_) return;
  const float* Xb = X + (long long)b * 64 * N_;
  float q[64], acc[64], nf[64];
  for (int c = 0; c < 64; ++c) { q[c] = Xb[(long long)c * N_ + n]; acc[c] = 0.f; }
  float wsum = 0.f;
  const int* row = knn64 + ((long long)b * N_ + n) * 64;
  for (int j = 0; j < 64; ++j) {
    int nb = row[j];
    float d2 = 0.f;
    for (int c = 0; c < 64; ++c) {
      float f = Xb[(long long)c * N_ + nb];
      nf[c] = f; float d = f - q[c]; d2 += d * d;
    }
    float w = expf(-d2 * 0.125f);          // 2*sigma^2 = 8
    if (w < 0.2f) w = 0.f;                  // tau
    wsum += w;
    if (w != 0.f) for (int c = 0; c < 64; ++c) acc[c] += w * nf[c];
  }
  float inv = 1.f / wsum, s = 0.f;
  for (int c = 0; c < 64; ++c) { float d = q[c] - acc[c] * inv; s += d * d; }
  score[(long long)b * N_ + n] = sqrtf(s);
}

// Per-batch bitonic sort on (score, idx); mode 0: desc (sharp), 1: asc (gentle).
__global__ __launch_bounds__(1024) void gdm_select(
    const float* __restrict__ score, int* __restrict__ out_idx, int mode,
    int N_, int M_)
{
  __shared__ float sv[NPTS];
  __shared__ int   si[NPTS];
  int b = blockIdx.x;
  for (int i = threadIdx.x; i < N_; i += 1024) {
    sv[i] = score[(long long)b * N_ + i]; si[i] = i;
  }
  __syncthreads();
  for (int size = 2; size <= N_; size <<= 1) {
    for (int stride = size >> 1; stride > 0; stride >>= 1) {
      for (int i = threadIdx.x; i < N_; i += 1024) {
        int j = i ^ stride;
        if (j > i) {
          float va = sv[i], vb = sv[j]; int ia = si[i], ib = si[j];
          bool dirUp = ((i & size) == 0);
          bool before = (mode == 0) ? (va > vb || (va == vb && ia < ib))
                                    : (va < vb || (va == vb && ia < ib));
          if (before != dirUp) { sv[i] = vb; si[i] = ib; sv[j] = va; si[j] = ia; }
        }
      }
      __syncthreads();
    }
  }
  for (int m = threadIdx.x; m < M_; m += 1024) out_idx[(long long)b * M_ + m] = si[m];
}

__global__ void gather_cols(const float* __restrict__ X, const int* __restrict__ idx,
                            float* __restrict__ outp, int C, int N_, int M_,
                            long long total)
{
  long long t = (long long)blockIdx.x * 256 + threadIdx.x;
  if (t >= total) return;
  int m = (int)(t % M_);
  long long bc = t / M_;          // b*C + c
  int b = (int)(bc / C);
  outp[t] = X[bc * N_ + idx[(long long)b * M_ + m]];
}

__global__ void copy_rows(const float* __restrict__ src, long long sbs,
                          float* __restrict__ dst, long long dbs,
                          int R, int N_, long long total)
{
  long long t = (long long)blockIdx.x * 256 + threadIdx.x;
  if (t >= total) return;
  int n = (int)(t % N_);
  long long br = t / N_;
  int b = (int)(br / R);
  int r = (int)(br - (long long)b * R);
  dst[(long long)b * dbs + (long long)r * N_ + n] =
      src[(long long)b * sbs + (long long)r * N_ + n];
}

__global__ void fill_f32(float* p, long long total, float v) {
  long long i = (long long)blockIdx.x * 256 + threadIdx.x;
  if (i < total) p[i] = v;
}

__global__ void fold_bn(const float* bias, const float* g, const float* bb,
                        const float* m, const float* v, float* sc, float* sh, int n) {
  int i = blockIdx.x * 256 + threadIdx.x;
  if (i >= n) return;
  float s = g[i] * rsqrtf(v[i] + BN_EPS);
  sc[i] = s;
  sh[i] = s * bias[i] + bb[i] - s * m[i];
}

__global__ void make_uv(const float* W, float* U, float* V, int O, int C) {
  int i = blockIdx.x * 256 + threadIdx.x;
  if (i >= O * C) return;
  int o = i / C, c = i - o * C;
  float wd = W[o * 2 * C + c];
  float wn = W[o * 2 * C + C + c];
  U[i] = wd + wn;
  V[i] = -wd;
}

__global__ __launch_bounds__(256) void pool_head(const float* __restrict__ h,
                                                 float* __restrict__ outp,
                                                 int C, int N_)
{
  __shared__ float smx[256], ssm[256];
  int b = blockIdx.y, o = blockIdx.x, tid = threadIdx.x;
  float mx = -3.4e38f, sm = 0.f;
  for (int n = tid; n < N_; n += 256) {
    float v = h[((long long)b * C + o) * N_ + n];
    mx = fmaxf(mx, v); sm += v;
  }
  smx[tid] = mx; ssm[tid] = sm;
  __syncthreads();
  for (int s = 128; s > 0; s >>= 1) {
    if (tid < s) { smx[tid] = fmaxf(smx[tid], smx[tid + s]); ssm[tid] += ssm[tid + s]; }
    __syncthreads();
  }
  if (tid == 0) {
    outp[(long long)b * 2 * C + o]     = smx[0];
    outp[(long long)b * 2 * C + C + o] = ssm[0] / (float)N_;
  }
}

// ---------------------------------------------------------------------------
// Host-side orchestration.
// ---------------------------------------------------------------------------
extern "C" void kernel_launch(void* const* d_in, const int* in_sizes, int n_in,
                              void* d_out, int out_size, void* d_ws, size_t ws_size,
                              hipStream_t stream) {
  (void)in_sizes; (void)n_in; (void)out_size; (void)ws_size;
  const float* x = (const float*)d_in[0];
  auto W = [&](int i) { return (const float*)d_in[i]; };
  float* ws = (float*)d_ws;
  size_t off = 0;
  auto alloc = [&](size_t n) { float* p = ws + off; off += n; return p; };

  const long long NP = NPTS;
  // folded affines
  float *sc1=alloc(64), *sh1=alloc(64), *sc11=alloc(64), *sh11=alloc(64);
  float *sc12=alloc(64), *sh12=alloc(64);
  float *sc2=alloc(64), *sh2=alloc(64), *sc21=alloc(64), *sh21=alloc(64);
  float *sc22=alloc(64), *sh22=alloc(64);
  float *sc3=alloc(128), *sh3=alloc(128), *sc31=alloc(128), *sh31=alloc(128);
  float *sc32=alloc(128), *sh32=alloc(128);
  float *sc4=alloc(512), *sh4=alloc(512);
  float *scw1s=alloc(64), *shw1s=alloc(64), *scw1g=alloc(64), *shw1g=alloc(64);
  float *scw2s=alloc(64), *shw2s=alloc(64), *scw2g=alloc(64), *shw2g=alloc(64);
  // edge U/V weights
  float *U1=alloc(64*3),    *V1=alloc(64*3);
  float *U2=alloc(64*67),   *V2=alloc(64*67);
  float *U3=alloc(128*131), *V3=alloc(128*131);
  // index / score buffers
  int *idx20 = (int*)alloc((size_t)BATCH*NP*KNN_K);
  int *idx64 = (int*)alloc((size_t)BATCH*NP*KG);
  int *idxS  = (int*)alloc((size_t)BATCH*MPTS);
  int *idxG  = (int*)alloc((size_t)BATCH*MPTS);
  float *score = alloc((size_t)BATCH*NP);
  // activations
  float *xs = alloc((size_t)BATCH*64*MPTS), *xg = alloc((size_t)BATCH*64*MPTS);
  float *gbuf = alloc((size_t)BATCH*32*MPTS), *phbuf = alloc((size_t)BATCH*32*MPTS);
  float *thbuf = alloc((size_t)BATCH*32*NP), *ybuf = alloc((size_t)BATCH*32*NP);
  float *Pb = alloc((size_t)BATCH*128*NP),  *Qb = alloc((size_t)BATCH*128*NP);
  float *x1 = alloc((size_t)BATCH*64*NP),  *x2 = alloc((size_t)BATCH*64*NP);
  float *x3 = alloc((size_t)BATCH*128*NP);
  float *buf67  = alloc((size_t)BATCH*67*NP);
  float *buf131 = alloc((size_t)BATCH*131*NP);
  float *cat128 = alloc((size_t)BATCH*128*NP);
  float *headb  = alloc((size_t)BATCH*256*NP);
  float *att    = alloc((size_t)BATCH*NP*MPTS);   // reused as head feature buffer

  auto fold = [&](int base, int n, float* sc, float* sh) {
    fold_bn<<<(n+255)/256, 256, 0, stream>>>(W(base+1), W(base+2), W(base+3),
                                             W(base+4), W(base+5), sc, sh, n);
  };
  auto fold_sg = [&](int ps, float* sc, float* sh) {
    fold_bn<<<1, 256, 0, stream>>>(W(ps+7), W(ps+8), W(ps+9), W(ps+10), W(ps+11),
                                   sc, sh, 64);
  };
  auto gemm = [&](const float* A, long long ar, long long ac, long long ab,
                  const float* Bm, long long br, long long bc, long long bb,
                  float* C, long long cr, long long cb,
                  const float* resid, long long rr, long long rb,
                  const float* sc, const float* sh,
                  int M, int Nc, int Kd, float alpha, int relu) {
    dim3 g((Nc + 63) / 64, (M + 63) / 64, BATCH);
    gemm_wmma<<<g, 128, 0, stream>>>(A, ar, ac, ab, Bm, br, bc, bb, C, cr, cb,
                                     resid, rr, rb, sc, sh, M, Nc, Kd, alpha, relu);
  };
  auto fill = [&](float* p, long long total, float v) {
    fill_f32<<<(unsigned)((total + 255) / 256), 256, 0, stream>>>(p, total, v);
  };

  // ---- prep: fold BN, build edge U/V ----
  fold(1, 64, sc1, sh1);   fold(7, 64, sc11, sh11);  fold(13, 64, sc12, sh12);
  fold(19, 64, sc2, sh2);  fold(25, 64, sc21, sh21); fold(31, 64, sc22, sh22);
  fold(37, 128, sc3, sh3); fold(43, 128, sc31, sh31); fold(49, 128, sc32, sh32);
  fold(55, 512, sc4, sh4);
  fold_sg(61, scw1s, shw1s); fold_sg(73, scw1g, shw1g);
  fold_sg(85, scw2s, shw2s); fold_sg(97, scw2g, shw2g);
  make_uv<<<(64*3+255)/256, 256, 0, stream>>>(W(1), U1, V1, 64, 3);
  make_uv<<<(64*67+255)/256, 256, 0, stream>>>(W(19), U2, V2, 64, 67);
  make_uv<<<(128*131+255)/256, 256, 0, stream>>>(W(37), U3, V3, 128, 131);

  auto edge_block = [&](const float* Xin, int Cin, const float* U, const float* V,
                        int O1, const float* sA1, const float* tA1,
                        const float* W2, const float* sA2, const float* tA2,
                        int O2, float* Xout) {
    knn_kernel<<<dim3(NPTS/256, BATCH), 256, 0, stream>>>(Xin, Cin, NPTS, KNN_K,
                                                          idx20, (long long)Cin*NP);
    gemm(U, Cin, 1, 0, Xin, NP, 1, (long long)Cin*NP, Pb, NP, (long long)O1*NP,
         0, 0, 0, 0, 0, O1, NPTS, Cin, 1.f, 0);
    gemm(V, Cin, 1, 0, Xin, NP, 1, (long long)Cin*NP, Qb, NP, (long long)O1*NP,
         0, 0, 0, 0, 0, O1, NPTS, Cin, 1.f, 0);
    fill(Xout, (long long)BATCH*O2*NP, 0.f);
    edge2_wmma<<<dim3((NPTS*KNN_K + 63)/64, BATCH), 128, 0, stream>>>(
        Pb, Qb, idx20, sA1, tA1, W2, sA2, tA2, Xout, NPTS, O1, O2, KNN_K);
  };
  auto gdm = [&](const float* Xf) {
    knn_kernel<<<dim3(NPTS/256, BATCH), 256, 0, stream>>>(Xf, 64, NPTS, KG, idx64,
                                                          (long long)64*NP);
    gdm_score<<<dim3(NPTS/256, BATCH), 256, 0, stream>>>(Xf, idx64, score, NPTS);
    gdm_select<<<BATCH, 1024, 0, stream>>>(score, idxS, 0, NPTS, MPTS);
    gdm_select<<<BATCH, 1024, 0, stream>>>(score, idxG, 1, NPTS, MPTS);
    long long tot = (long long)BATCH*64*MPTS;
    gather_cols<<<(unsigned)((tot+255)/256), 256, 0, stream>>>(Xf, idxS, xs, 64, NPTS, MPTS, tot);
    gather_cols<<<(unsigned)((tot+255)/256), 256, 0, stream>>>(Xf, idxG, xg, 64, NPTS, MPTS, tot);
  };
  auto sgcam = [&](const float* xq, const float* xv, int ps,
                   const float* scw, const float* shw, float* outp) {
    gemm(W(ps+0), 64,1,0, xv, MPTS,1,(long long)64*MPTS, gbuf, MPTS,(long long)32*MPTS,
         0,0,0, 0, W(ps+1), 32, MPTS, 64, 1.f, 0);                        // g
    gemm(W(ps+2), 64,1,0, xq, NP,1,(long long)64*NP, thbuf, NP,(long long)32*NP,
         0,0,0, 0, W(ps+3), 32, NPTS, 64, 1.f, 0);                        // theta
    gemm(W(ps+4), 64,1,0, xv, MPTS,1,(long long)64*MPTS, phbuf, MPTS,(long long)32*MPTS,
         0,0,0, 0, W(ps+5), 32, MPTS, 64, 1.f, 0);                        // phi
    gemm(thbuf, 1, NP, (long long)32*NP, phbuf, MPTS,1,(long long)32*MPTS,
         att, MPTS, (long long)NP*MPTS, 0,0,0, 0,0,
         NPTS, MPTS, 32, 1.f/(float)MPTS, 0);                             // att = th^T ph / M
    gemm(gbuf, MPTS,1,(long long)32*MPTS, att, 1, MPTS, (long long)NP*MPTS,
         ybuf, NP, (long long)32*NP, 0,0,0, 0,0, 32, NPTS, MPTS, 1.f, 0); // y = g att^T
    gemm(W(ps+6), 32,1,0, ybuf, NP,1,(long long)32*NP, outp, NP,(long long)128*NP,
         xq, NP, (long long)64*NP, scw, shw, 64, NPTS, 32, 1.f, 0);       // bn(Wy)+x
  };

  // ---------------- block 1 ----------------
  edge_block(x, 3, U1, V1, 64, sc1, sh1, W(7), sc11, sh11, 64, x1);
  gdm(x1);
  sgcam(x1, xs, 61, scw1s, shw1s, cat128);
  sgcam(x1, xg, 73, scw1g, shw1g, cat128 + 64*NP);
  gemm(W(13), 128,1,0, cat128, NP,1,(long long)128*NP, headb, NP,(long long)256*NP,
       0,0,0, sc12, sh12, 64, NPTS, 128, 1.f, 1);                          // z1 -> head[0:64]
  { long long t3 = (long long)BATCH*3*NP, t64 = (long long)BATCH*64*NP;
    copy_rows<<<(unsigned)((t3+255)/256), 256, 0, stream>>>(x, 3*NP, buf67, 67*NP, 3, NPTS, t3);
    copy_rows<<<(unsigned)((t64+255)/256), 256, 0, stream>>>(headb, 256*NP, buf67 + 3*NP, 67*NP, 64, NPTS, t64); }

  // ---------------- block 2 ----------------
  edge_block(buf67, 67, U2, V2, 64, sc2, sh2, W(25), sc21, sh21, 64, x2);
  gdm(x2);
  sgcam(x2, xs, 85, scw2s, shw2s, cat128);
  sgcam(x2, xg, 97, scw2g, shw2g, cat128 + 64*NP);
  gemm(W(31), 128,1,0, cat128, NP,1,(long long)128*NP, headb + 64*NP, NP,(long long)256*NP,
       0,0,0, sc22, sh22, 64, NPTS, 128, 1.f, 1);                          // z2 -> head[64:128]
  { long long t67 = (long long)BATCH*67*NP, t64 = (long long)BATCH*64*NP;
    copy_rows<<<(unsigned)((t67+255)/256), 256, 0, stream>>>(buf67, 67*NP, buf131, 131*NP, 67, NPTS, t67);
    copy_rows<<<(unsigned)((t64+255)/256), 256, 0, stream>>>(headb + 64*NP, 256*NP, buf131 + 67*NP, 131*NP, 64, NPTS, t64); }

  // ---------------- block 3 ----------------
  edge_block(buf131, 131, U3, V3, 128, sc3, sh3, W(43), sc31, sh31, 128, x3);
  gemm(W(49), 128,1,0, x3, NP,1,(long long)128*NP, headb + 128*NP, NP,(long long)256*NP,
       0,0,0, sc32, sh32, 128, NPTS, 128, 1.f, 1);                         // z3 -> head[128:256]

  // ---------------- head ----------------
  gemm(W(55), 256,1,0, headb, NP,1,(long long)256*NP, att, NP,(long long)512*NP,
       0,0,0, sc4, sh4, 512, NPTS, 256, 1.f, 1);                           // h (reuse att buf)
  pool_head<<<dim3(512, BATCH), 256, 0, stream>>>(att, (float*)d_out, 512, NPTS);
}
